// SelfAttentionLayer_60112362274836
// MI455X (gfx1250) — matrix operand order repaired and checked by reading, not compiled, not run on hardware
//
#include <hip/hip_runtime.h>
#include <hip/hip_bf16.h>

// N=2048, IN_DIM=128, OUT_DIM=64, alpha=0.2
// lrelu(x) = C1*x + C2*|x|, C1=(1+a)/2=0.6, C2=(1-a)/2=0.4

typedef __bf16 bf16_t;
typedef __attribute__((ext_vector_type(16))) __bf16 v16bf;
typedef __attribute__((ext_vector_type(8)))  __bf16 v8bf;
typedef __attribute__((ext_vector_type(8)))  float  v8f;
typedef unsigned int u32x4 __attribute__((ext_vector_type(4)));
typedef int          i32x4 __attribute__((ext_vector_type(4)));
typedef int          i32x8 __attribute__((ext_vector_type(8)));

#define NTOT 2048
#define KDIM 128
#define DDIM 64
#define NROW 16           // rows of output per workgroup in kernel 2
#define PPAD 2052         // 2048 + 4 floats pad (bank-conflict-free WMMA A reads)
#define WPAD 68           // 64 + 4 floats pad for whi tile (matches TDM pad descriptor)

// ---------------------------------------------------------------------------
// Kernel 1: Wh = H @ W^T  via v_wmma_f32_16x16x32_bf16.
// Grid 128 WGs x 128 thr (4 waves). Wave w computes the 16x16 tile at
// (m0 = 16*blockIdx, n0 = 16*w).  K = 128 -> 4 WMMA steps of K=32.
// Also emits Wh^T in bf16 (row-major [64][2048]) for the P@Wh B-operand.
// ---------------------------------------------------------------------------
__global__ __launch_bounds__(128)
void wh_gemm_kernel(const float* __restrict__ H, const float* __restrict__ W,
                    float* __restrict__ Wh, bf16_t* __restrict__ WhT) {
  const int tid   = threadIdx.x;
  const int wave  = tid >> 5;
  const int lane  = tid & 31;
  const int mlane = lane & 15;
  const int h     = lane >> 4;
  const int m0    = blockIdx.x * 16;
  const int n0    = wave * 16;

  v8f c = {};
  #pragma unroll
  for (int k0 = 0; k0 < KDIM; k0 += 32) {
    // A fragment: H[m0+mlane][k0 + h*16 + j] (j=0..15)
    const float4* ar = (const float4*)(H + (m0 + mlane) * KDIM + k0 + h * 16);
    float4 a0 = ar[0], a1 = ar[1], a2 = ar[2], a3 = ar[3];
    v16bf af;
    af[0]=(__bf16)a0.x; af[1]=(__bf16)a0.y; af[2]=(__bf16)a0.z; af[3]=(__bf16)a0.w;
    af[4]=(__bf16)a1.x; af[5]=(__bf16)a1.y; af[6]=(__bf16)a1.z; af[7]=(__bf16)a1.w;
    af[8]=(__bf16)a2.x; af[9]=(__bf16)a2.y; af[10]=(__bf16)a2.z; af[11]=(__bf16)a2.w;
    af[12]=(__bf16)a3.x; af[13]=(__bf16)a3.y; af[14]=(__bf16)a3.z; af[15]=(__bf16)a3.w;
    // B fragment: B[k][n]=W[n][k]; lane holds column n=n0+mlane, K=k0+h*16+0..15
    const float4* br = (const float4*)(W + (n0 + mlane) * KDIM + k0 + h * 16);
    v16bf bf;
    #pragma unroll
    for (int q = 0; q < 4; ++q) {
      float4 bv = br[q];
      bf[4*q+0]=(__bf16)bv.x; bf[4*q+1]=(__bf16)bv.y;
      bf[4*q+2]=(__bf16)bv.z; bf[4*q+3]=(__bf16)bv.w;
    }
    c = __builtin_amdgcn_wmma_f32_16x16x32_bf16(false, af, false, bf,
                                                (short)0, c, false, false);
  }
  const int n = n0 + mlane;
  #pragma unroll
  for (int r = 0; r < 8; ++r) {
    const int m = m0 + h * 8 + r;
    Wh[m * DDIM + n]  = c[r];
    WhT[n * NTOT + m] = (__bf16)c[r];
  }
}

// LDS byte offset of a __shared__ object (addrspacecast generic->AS3, ptrtoint)
__device__ __forceinline__ unsigned lds_offset(const void* p) {
  return (unsigned)(unsigned long long)
         (__attribute__((address_space(3))) const void*)p;
}

// ---------------------------------------------------------------------------
// Kernel 2: fused  e -> softmax -> attention @ Wh  for a 16-row block.
// Grid 128 WGs x 256 thr (8 waves). The full 16x2048 attention-row tile
// lives in LDS (~140 KB of CDNA5's 320 KB) so e is never spilled to HBM.
// Prologue staging uses the CDNA5 DMA paths: TDM tensor_load_to_lds for the
// Wh row-tile (with hardware row padding) and global_load_async_to_lds_b128
// for the attention vector.
// ---------------------------------------------------------------------------
__global__ __launch_bounds__(256)
void attn_kernel(const float* __restrict__ a_vec,
                 const float* __restrict__ Wh,
                 const bf16_t* __restrict__ WhT,
                 float* __restrict__ out) {
  __shared__ alignas(16) float pbuf[NROW][PPAD];   // e, then P (normalized)
  __shared__ alignas(16) float whi[NROW][WPAD];    // this block's Wh rows
  __shared__ alignas(16) float araw[DDIM];         // raw a vector
  __shared__ alignas(16) float alin[DDIM];         // 0.6*a
  __shared__ alignas(16) float babs[DDIM];         // 0.4*a
  __shared__ float s1loc[NROW];
  __shared__ float cred[4][16][16];                // K-split partial C

  const int tid  = threadIdx.x;
  const int i0   = blockIdx.x * NROW;
  const int wave = tid >> 5;
  const int lane = tid & 31;

  // ---- prologue: DMA-stage whi tile (TDM) and a vector (async-to-LDS) ----
#if __has_builtin(__builtin_amdgcn_tensor_load_to_lds)
  if (wave == 0) {
    // D# group 0: count=1 | lds_addr | global_addr | type=2
    const unsigned long long ga =
        (unsigned long long)(const void*)(Wh + (size_t)i0 * DDIM);
    u32x4 g0;
    g0[0] = 1u;                                   // count=1 (valid descriptor)
    g0[1] = lds_offset(&whi[0][0]);               // lds_addr (bytes)
    g0[2] = (unsigned)(ga & 0xFFFFFFFFu);         // global_addr[31:0]
    g0[3] = (unsigned)((ga >> 32) & 0x01FFFFFFu)  // global_addr[56:32]
          | (2u << 30);                           // type=2 ("image")
    // D# group 1: data_size=4B, pad 4 dwords after every 64 dwords,
    // tensor 64 x 16 (stride 64), tile 64 x 16.
    i32x8 g1;
    g1[0] = (2 << 16)      // data_size = 2 -> 4 bytes
          | (1 << 20)      // pad_enable
          | (5 << 22)      // pad_interval code 5 -> every 64 DWORDs
          | (3 << 25);     // pad_amount  code 3 -> 4 DWORDs
    g1[1] = (DDIM & 0xFFFF) << 16;                // tensor_dim0 = 64 (lo16)
    g1[2] = (NROW & 0xFFFF) << 16;                // tensor_dim1 = 16 (lo16)
    g1[3] = (DDIM & 0xFFFF) << 16;                // tile_dim0 = 64
    g1[4] = NROW;                                 // tile_dim1 = 16, tile_dim2 = 0
    g1[5] = DDIM;                                 // tensor_dim0_stride = 64 (lo32)
    g1[6] = 0;
    g1[7] = 0;
    const i32x4 gz = {};
# if __has_include(<hip/amd_detail/amd_gfx1250_TDM.h>)
    const i32x8 gz8 = {};
    __builtin_amdgcn_tensor_load_to_lds(g0, g1, gz, gz, gz8, 0);
# else
    __builtin_amdgcn_tensor_load_to_lds(g0, g1, gz, gz, 0);
# endif
    __builtin_amdgcn_s_wait_tensorcnt(0);
  }
#else
  for (int idx = tid; idx < NROW * DDIM; idx += 256) {
    const int i = idx >> 6, d = idx & 63;
    whi[i][d] = Wh[(i0 + i) * DDIM + d];
  }
#endif
  if (wave == 1) {
    if (lane < 16) {  // 16 lanes x 16B = 256B = whole a vector
      const unsigned ldst = lds_offset(&araw[lane * 4]);
      const float*   gsrc = a_vec + lane * 4;
      asm volatile("global_load_async_to_lds_b128 %0, %1, off"
                   :: "v"(ldst), "v"(gsrc) : "memory");
    }
    asm volatile("s_wait_asynccnt 0x0" ::: "memory");
  }
  __syncthreads();

  if (tid < DDIM) {
    const float av = araw[tid];
    alin[tid] = 0.6f * av;
    babs[tid] = 0.4f * av;
  }
  __syncthreads();
  if (tid < NROW) {
    float s = 0.f;
    for (int d = 0; d < DDIM; ++d) s += alin[d] * whi[tid][d];
    s1loc[tid] = s;                                 // 0.6 * (a . Wh[i])
  }
  __syncthreads();

  // ---- phase 1: e[i][j] for all 2048 j (each thread owns 8 columns) ----
  const float4* alin4 = (const float4*)alin;
  const float4* babs4 = (const float4*)babs;
  for (int jb = 0; jb < NTOT; jb += 256) {
    const int j = jb + tid;
    if (jb + 256 < NTOT) __builtin_prefetch(Wh + (j + 256) * DDIM, 0, 0);
    float4 whjv[16];
    const float4* wr = (const float4*)(Wh + j * DDIM);
    #pragma unroll
    for (int q = 0; q < 16; ++q) whjv[q] = wr[q];

    float acc[NROW];
    #pragma unroll
    for (int i = 0; i < NROW; ++i) acc[i] = 0.f;
    float sj = 0.f;

    #pragma unroll
    for (int d4 = 0; d4 < 16; ++d4) {
      const float4 t  = whjv[d4];
      const float4 al = alin4[d4];                  // LDS broadcast (uniform)
      const float4 bb = babs4[d4];
      sj += al.x * t.x + al.y * t.y + al.z * t.z + al.w * t.w;
      #pragma unroll
      for (int i = 0; i < NROW; ++i) {
        const float4 wv = *(const float4*)&whi[i][d4 * 4];  // uniform broadcast
        acc[i] += bb.x * __builtin_fabsf(wv.x + t.x)
                + bb.y * __builtin_fabsf(wv.y + t.y)
                + bb.z * __builtin_fabsf(wv.z + t.z)
                + bb.w * __builtin_fabsf(wv.w + t.w);
      }
    }
    #pragma unroll
    for (int i = 0; i < NROW; ++i)
      pbuf[i][j] = acc[i] + s1loc[i] + sj;
  }
  __syncthreads();

  // ---- phase 2: row softmax in LDS (one wave handles 2 rows) ----
  for (int rr = 0; rr < 2; ++rr) {
    const int row = wave * 2 + rr;
    float m = -3.402823e38f;
    for (int j = lane; j < NTOT; j += 32) m = fmaxf(m, pbuf[row][j]);
    #pragma unroll
    for (int off = 16; off > 0; off >>= 1) m = fmaxf(m, __shfl_xor(m, off, 32));
    float s = 0.f;
    for (int j = lane; j < NTOT; j += 32) {
      const float p = __expf(pbuf[row][j] - m);
      pbuf[row][j] = p;
      s += p;
    }
    #pragma unroll
    for (int off = 16; off > 0; off >>= 1) s += __shfl_xor(s, off, 32);
    const float inv = 1.0f / s;
    for (int j = lane; j < NTOT; j += 32) pbuf[row][j] *= inv;
  }
  __syncthreads();

  // ---- phase 3: out = P @ Wh via WMMA. 8 waves = 4 N-tiles x 2 K-halves ----
  const int nt    = wave & 3;        // which 16-col tile of the 64 outputs
  const int kh    = wave >> 2;       // K half (0: 0..1023, 1: 1024..2047)
  const int mlane = lane & 15;
  const int h     = lane >> 4;
  v8f c = {};
  const int kbase = kh * (NTOT / 2);
  for (int ks = 0; ks < NTOT / 2; ks += 32) {
    const int k0 = kbase + ks;
    // A fragment from P (fp32 LDS -> bf16)
    const float4* pr = (const float4*)&pbuf[mlane][k0 + h * 16];
    float4 a0 = pr[0], a1 = pr[1], a2 = pr[2], a3 = pr[3];
    v16bf af;
    af[0]=(__bf16)a0.x; af[1]=(__bf16)a0.y; af[2]=(__bf16)a0.z; af[3]=(__bf16)a0.w;
    af[4]=(__bf16)a1.x; af[5]=(__bf16)a1.y; af[6]=(__bf16)a1.z; af[7]=(__bf16)a1.w;
    af[8]=(__bf16)a2.x; af[9]=(__bf16)a2.y; af[10]=(__bf16)a2.z; af[11]=(__bf16)a2.w;
    af[12]=(__bf16)a3.x; af[13]=(__bf16)a3.y; af[14]=(__bf16)a3.z; af[15]=(__bf16)a3.w;
    // B fragment: WhT[n][k0 + h*16 + 0..15], contiguous bf16 -> two b128 loads
    const v8bf* bp = (const v8bf*)(WhT + (nt * 16 + mlane) * NTOT + k0 + h * 16);
    const v8bf blo = bp[0], bhi = bp[1];
    v16bf bf;
    #pragma unroll
    for (int q = 0; q < 8; ++q) { bf[q] = blo[q]; bf[8 + q] = bhi[q]; }
    c = __builtin_amdgcn_wmma_f32_16x16x32_bf16(false, af, false, bf,
                                                (short)0, c, false, false);
  }
  if (kh == 1) {
    #pragma unroll
    for (int r = 0; r < 8; ++r) cred[nt][h * 8 + r][mlane] = c[r];
  }
  __syncthreads();
  if (kh == 0) {
    #pragma unroll
    for (int r = 0; r < 8; ++r) {
      const int m = i0 + h * 8 + r;
      out[m * DDIM + nt * 16 + mlane] = c[r] + cred[nt][h * 8 + r][mlane];
    }
  }
}

// ---------------------------------------------------------------------------
extern "C" void kernel_launch(void* const* d_in, const int* in_sizes, int n_in,
                              void* d_out, int out_size, void* d_ws, size_t ws_size,
                              hipStream_t stream) {
  const float* H = (const float*)d_in[0];   // [2048][128]
  const float* W = (const float*)d_in[1];   // [64][128]
  const float* a = (const float*)d_in[2];   // [64]
  float* out = (float*)d_out;               // [2048][64]

  float*  Wh  = (float*)d_ws;                                  // 512 KB
  bf16_t* WhT = (bf16_t*)((char*)d_ws + (size_t)NTOT * DDIM * sizeof(float)); // 256 KB

  wh_gemm_kernel<<<dim3(NTOT / 16), dim3(128), 0, stream>>>(H, W, Wh, WhT);
  attn_kernel<<<dim3(NTOT / NROW), dim3(256), 0, stream>>>(a, Wh, WhT, out);
}
